// Embedder_89687507076271
// MI455X (gfx1250) — matrix operand order repaired and checked by reading, not compiled, not run on hardware
//
#include <hip/hip_runtime.h>
#include <cmath>

typedef _Float16 v16h __attribute__((ext_vector_type(16)));
typedef _Float16 h2v  __attribute__((ext_vector_type(2)));
typedef float    v8f  __attribute__((ext_vector_type(8)));
typedef float    v2f  __attribute__((ext_vector_type(2)));

#define N_ROWS      262144
#define D_OUT       512
#define ROWS_PER_WG 256
#define THREADS     512
#define WAVES       16

// ---- LDS layout (bytes) ----
// Weights staged as f16 in "B-operand swizzled" order:
//   chunk id = ct*KC + kc; each chunk = 512 halves [lane][16 halves];
//   lane = hi*16 + (feat%16); halves = W[feat][kc*32 + hi*16 + 0..15]
#define OFF_W2  0         // 256x128 -> 64 chunks * 1KB = 64KB
#define OFF_W3  65536     // 128x256 -> 64 chunks      = 64KB
#define OFF_W4  131072    // 512x128 -> 128 chunks     = 128KB
#define OFF_W1  262144    // 128x4 f32 (2KB)
#define OFF_B1  264192    // 128 f32
#define OFF_B2  264704    // 256 f32
#define OFF_B3  265728    // 128 f32
#define OFF_SCR 266240    // 16 waves * 1KB streaming A-chunk scratch
#define SMEM_BYTES 282624

// Branchless exact-GELU: erf via Abramowitz-Stegun 7.1.26 (|err|<=1.5e-7),
// v_rcp + v_exp transcendentals, no EXEC divergence.
__device__ __forceinline__ float gelu_exact(float x) {
    const float z = __builtin_fabsf(x) * 0.7071067811865476f;     // |x|/sqrt(2)
    const float t = __builtin_amdgcn_rcpf(__builtin_fmaf(0.3275911f, z, 1.0f));
    float p = __builtin_fmaf(1.061405429f, t, -1.453152027f);
    p = __builtin_fmaf(p, t, 1.421413741f);
    p = __builtin_fmaf(p, t, -0.284496736f);
    p = __builtin_fmaf(p, t, 0.254829592f);
    p *= t;
    const float e  = __builtin_amdgcn_exp2f(z * z * -1.4426950408889634f); // e^{-z^2}
    const float er = __builtin_fmaf(-p, e, 1.0f);                 // erf(z), z>=0
    const float erfv = __builtin_copysignf(er, x);
    return 0.5f * x * (1.0f + erfv);
}

// Convert f32 weights [F][Ktot] (row-major) -> f16 B-operand-swizzled LDS buffer.
__device__ __forceinline__ void stage_weight(const float* __restrict__ src,
                                             _Float16* dst, int F, int Ktot, int tid) {
    const int KC    = Ktot >> 5;
    const int pairs = (F * Ktot) >> 1;
    for (int p = tid; p < pairs; p += THREADS) {
        const int d     = p << 1;          // destination half index
        const int kk    = d & 15;          // even
        const int lane  = (d >> 4) & 31;
        const int chunk = d >> 9;
        const int kc    = chunk & (KC - 1);
        const int ct    = chunk / KC;
        const int fl    = lane & 15, hi = lane >> 4;
        const int feat  = ct * 16 + fl;
        const int k     = kc * 32 + hi * 16 + kk;
        const float2 s  = *(const float2*)(src + feat * Ktot + k);
        h2v h; h.x = (_Float16)s.x; h.y = (_Float16)s.y;
        *(h2v*)(dst + d) = h;
    }
}

// Store one f32 D-tile element (row-in-tile M, kin = k%32) into the wave's
// 1KB A-operand-swizzled f16 streaming chunk.
__device__ __forceinline__ void scr_store(_Float16* scr, int kin, int row_in_tile, float v) {
    const int g    = kin >> 4;
    const int slot = kin & 15;
    scr[(g * 16 + row_in_tile) * 16 + slot] = (_Float16)v;
}

__global__ void __launch_bounds__(THREADS, 1)
embedder_kernel(const float* __restrict__ joint_info,
                const int*   __restrict__ joint_token,
                const float* __restrict__ emb,
                const float* __restrict__ W1, const float* __restrict__ b1,
                const float* __restrict__ W2, const float* __restrict__ b2,
                const float* __restrict__ W3, const float* __restrict__ b3,
                const float* __restrict__ W4,
                float* __restrict__ out) {
    extern __shared__ char smem[];
    _Float16* w2h = (_Float16*)(smem + OFF_W2);
    _Float16* w3h = (_Float16*)(smem + OFF_W3);
    _Float16* w4h = (_Float16*)(smem + OFF_W4);
    float*    w1f = (float*)(smem + OFF_W1);
    float*    b1f = (float*)(smem + OFF_B1);
    float*    b2f = (float*)(smem + OFF_B2);
    float*    b3f = (float*)(smem + OFF_B3);

    const int tid = threadIdx.x;

    // ---- cooperative weight staging (f32 -> f16, WMMA-operand order) ----
    stage_weight(W2, w2h, 256, 128, tid);
    stage_weight(W3, w3h, 128, 256, tid);
    stage_weight(W4, w4h, 512, 128, tid);
    for (int i = tid; i < 512; i += THREADS) w1f[i] = W1[i];
    for (int i = tid; i < 128; i += THREADS) b1f[i] = b1[i];
    for (int i = tid; i < 256; i += THREADS) b2f[i] = b2[i];
    for (int i = tid; i < 128; i += THREADS) b3f[i] = b3[i];
    __syncthreads();

    const int lane = tid & 31;
    const int wave = tid >> 5;
    const int fl   = lane & 15;   // N / row selector within half-wave
    const int hi   = lane >> 4;   // half-wave selector
    _Float16* scr  = (_Float16*)(smem + OFF_SCR) + wave * 512;   // 1KB per wave

    const int row_base = blockIdx.x * ROWS_PER_WG + wave * 16;

    // ---- per-tile pad mask + token, broadcast via wave32 shuffle ----
    int padi, tok;
    {
        const float4 ji = *(const float4*)(joint_info + (size_t)(row_base + fl) * 4);
        padi = (ji.x == 0.0f && ji.y == 0.0f && ji.z == 0.0f && ji.w == 0.0f) ? 1 : 0;
        tok  = joint_token[row_base + fl];
    }
    int padr[8], tokr[8];
    #pragma unroll
    for (int r = 0; r < 8; ++r) {
        padr[r] = __shfl(padi, hi * 8 + r);
        tokr[r] = __shfl(tok,  hi * 8 + r);
    }

    // ================= Layer 1: [16x4] @ [4x128]^T, exact f32 WMMA (K=4) ====
    // Stream: each (even,odd) ct pair fills one 32-deep A chunk -> registers.
    v2f a1;
    {
        const float2 t = *(const float2*)(joint_info + (size_t)(row_base + fl) * 4 + hi * 2);
        a1[0] = t.x; a1[1] = t.y;
    }
    v16h a2[4];
    #pragma unroll
    for (int ct = 0; ct < 8; ++ct) {
        const int feat = ct * 16 + fl;
        v2f bop;
        { const float2 t = *(const float2*)(w1f + feat * 4 + hi * 2); bop[0] = t.x; bop[1] = t.y; }
        v8f c = {};
        c = __builtin_amdgcn_wmma_f32_16x16x4_f32(false, a1, false, bop, (short)0, c, false, false);
        const float bias = b1f[feat];
        const int kin = ((ct & 1) << 4) | fl;
        #pragma unroll
        for (int r = 0; r < 8; ++r)
            scr_store(scr, kin, hi * 8 + r, gelu_exact(c[r] + bias));
        if (ct & 1) a2[ct >> 1] = *(const v16h*)(scr + lane * 16);
    }

    // ================= Layer 2: h1 @ W2^T -> h2 (16x256), f16 WMMA =========
    v16h a3[8];
    #pragma unroll
    for (int ct = 0; ct < 16; ++ct) {
        const int feat = ct * 16 + fl;
        const float bias = b2f[feat];
        v8f acc = {bias, bias, bias, bias, bias, bias, bias, bias};
        #pragma unroll
        for (int kc = 0; kc < 4; ++kc) {
            const v16h bop = *(const v16h*)(w2h + ((ct * 4 + kc) * 512 + lane * 16));
            acc = __builtin_amdgcn_wmma_f32_16x16x32_f16(false, a2[kc], false, bop,
                                                         (short)0, acc, false, false);
        }
        const int kin = ((ct & 1) << 4) | fl;
        #pragma unroll
        for (int r = 0; r < 8; ++r)
            scr_store(scr, kin, hi * 8 + r, gelu_exact(acc[r]));
        if (ct & 1) a3[ct >> 1] = *(const v16h*)(scr + lane * 16);
    }

    // ================= Layer 3: h2 @ W3^T -> h3 (16x128) ====================
    v16h a4[4];
    #pragma unroll
    for (int ct = 0; ct < 8; ++ct) {
        const int feat = ct * 16 + fl;
        const float bias = b3f[feat];
        v8f acc = {bias, bias, bias, bias, bias, bias, bias, bias};
        #pragma unroll
        for (int kc = 0; kc < 8; ++kc) {
            const v16h bop = *(const v16h*)(w3h + ((ct * 8 + kc) * 512 + lane * 16));
            acc = __builtin_amdgcn_wmma_f32_16x16x32_f16(false, a3[kc], false, bop,
                                                         (short)0, acc, false, false);
        }
        const int kin = ((ct & 1) << 4) | fl;
        #pragma unroll
        for (int r = 0; r < 8; ++r)
            scr_store(scr, kin, hi * 8 + r, gelu_exact(acc[r]));
        if (ct & 1) a4[ct >> 1] = *(const v16h*)(scr + lane * 16);
    }

    // ===== Layer 4: h3 @ W4^T (no bias) + emb[token], pad -> emb[0] =========
    for (int ct = 0; ct < 32; ++ct) {
        v8f acc = {};
        #pragma unroll
        for (int kc = 0; kc < 4; ++kc) {
            const v16h bop = *(const v16h*)(w4h + ((ct * 4 + kc) * 512 + lane * 16));
            acc = __builtin_amdgcn_wmma_f32_16x16x32_f16(false, a4[kc], false, bop,
                                                         (short)0, acc, false, false);
        }
        const int col = ct * 16 + fl;
        const float e0 = emb[col];           // emb row 0 (pad rows)
        #pragma unroll
        for (int r = 0; r < 8; ++r) {
            const int row = row_base + hi * 8 + r;
            const float ev = emb[(size_t)tokr[r] * D_OUT + col];
            const float v  = padr[r] ? e0 : (acc[r] + ev);
            out[(size_t)row * D_OUT + col] = v;
        }
    }
}

extern "C" void kernel_launch(void* const* d_in, const int* in_sizes, int n_in,
                              void* d_out, int out_size, void* d_ws, size_t ws_size,
                              hipStream_t stream) {
    const float* joint_info  = (const float*)d_in[0];
    const int*   joint_token = (const int*)  d_in[1];
    const float* emb = (const float*)d_in[2];
    const float* W1  = (const float*)d_in[3];
    const float* b1  = (const float*)d_in[4];
    const float* W2  = (const float*)d_in[5];
    const float* b2  = (const float*)d_in[6];
    const float* W3  = (const float*)d_in[7];
    const float* b3  = (const float*)d_in[8];
    const float* W4  = (const float*)d_in[9];
    float* out = (float*)d_out;

    (void)hipFuncSetAttribute((const void*)embedder_kernel,
                              hipFuncAttributeMaxDynamicSharedMemorySize, SMEM_BYTES);
    dim3 grid(N_ROWS / ROWS_PER_WG);
    embedder_kernel<<<grid, THREADS, SMEM_BYTES, stream>>>(
        joint_info, joint_token, emb, W1, b1, W2, b2, W3, b3, W4, out);
}
